// SelectiveSSMBlock_20126216749349
// MI455X (gfx1250) — matrix operand order repaired and checked by reading, not compiled, not run on hardware
//
#include <hip/hip_runtime.h>
#include <hip/hip_bf16.h>
#include <stdint.h>

// ---------------------------------------------------------------------------
// Problem dims
// ---------------------------------------------------------------------------
#define D_MODEL 768
#define D_STATE 64
#define D_CONV  4
#define D_INNER 1536          // 768*2
#define BATCH   2
#define SEQ     1024
#define ROWS    (BATCH*SEQ)   // 2048
#define NPROJ   (2*D_STATE+1) // 129
#define LN_EPS  1e-5f

typedef __attribute__((ext_vector_type(16))) _Float16 v16h;
typedef __attribute__((ext_vector_type(8)))  _Float16 v8h;
typedef __attribute__((ext_vector_type(8)))  float    v8f;

// ---------------------------------------------------------------------------
// 1) LayerNorm -> f16 activations (row-major ROWS x D_MODEL)
// ---------------------------------------------------------------------------
__global__ __launch_bounds__(256)
void ssm_ln_kernel(const float* __restrict__ x,
                   const float* __restrict__ w,
                   const float* __restrict__ b,
                   _Float16* __restrict__ h)
{
    const int row = blockIdx.x;
    const int tid = threadIdx.x;
    const float* xr = x + (size_t)row * D_MODEL;
    __shared__ float red[256];

    float s = 0.f;
    for (int k = tid; k < D_MODEL; k += 256) s += xr[k];
    red[tid] = s; __syncthreads();
    for (int off = 128; off > 0; off >>= 1) {
        if (tid < off) red[tid] += red[tid + off];
        __syncthreads();
    }
    const float mu = red[0] * (1.f / D_MODEL);
    __syncthreads();

    float v = 0.f;
    for (int k = tid; k < D_MODEL; k += 256) { float d = xr[k] - mu; v += d * d; }
    red[tid] = v; __syncthreads();
    for (int off = 128; off > 0; off >>= 1) {
        if (tid < off) red[tid] += red[tid + off];
        __syncthreads();
    }
    const float rstd = rsqrtf(red[0] * (1.f / D_MODEL) + LN_EPS);

    for (int k = tid; k < D_MODEL; k += 256)
        h[(size_t)row * D_MODEL + k] = (_Float16)((xr[k] - mu) * rstd * w[k] + b[k]);
}

// ---------------------------------------------------------------------------
// 2) Transpose-convert weights: f32 (K x N) -> f16 (N x K)   (for B fragments)
// ---------------------------------------------------------------------------
__global__ __launch_bounds__(256)
void ssm_wt_kernel(const float* __restrict__ in, _Float16* __restrict__ out,
                   int K, int N)
{
    int idx = blockIdx.x * 256 + threadIdx.x;
    if (idx < K * N) {
        int k = idx / N, n = idx % N;
        out[(size_t)n * K + k] = (_Float16)in[idx];
    }
}

// ---------------------------------------------------------------------------
// 3) WMMA GEMM: C[M,N] = A[M,K] * Bt[N,K]^T  (+ optional residual), f16 in,
//    f32 accumulate.  Register-blocked: one wave computes a 32x64 output tile
//    (2 M-tiles x 4 N-tiles) -> 8 WMMAs per K=32 step, A frags reused 4x.
//    block = (32,4): 4 waves cover 4 consecutive 64-wide N groups.
// ---------------------------------------------------------------------------
__global__ __launch_bounds__(128)
void ssm_wmma_gemm_kernel(const _Float16* __restrict__ A,
                          const _Float16* __restrict__ Bt,
                          float* __restrict__ C,
                          const float* __restrict__ residual, // may be null
                          int M, int N, int K)
{
    const int lane   = threadIdx.x;                 // 0..31 (wave32)
    const int mtile0 = blockIdx.x * 2;              // two 16-row M tiles
    const int ntile0 = (blockIdx.y * 4 + threadIdx.y) * 4; // four 16-col N tiles

    const int mr  = (lane & 15);
    const int nr  = (lane & 15);
    const int kbA = (lane >> 4) * 16;               // A: K base per lane half
    const int kbB = (lane >> 4) * 16;               // B: K base per lane half

    const _Float16* arow0 = A + (size_t)(mtile0 * 16 + mr) * K;
    const _Float16* arow1 = arow0 + (size_t)16 * K;
    const _Float16* brow[4];
    #pragma unroll
    for (int j = 0; j < 4; ++j)
        brow[j] = Bt + (size_t)((ntile0 + j) * 16 + nr) * K;

    v8f acc[2][4];
    #pragma unroll
    for (int i = 0; i < 2; ++i)
        #pragma unroll
        for (int j = 0; j < 4; ++j) acc[i][j] = (v8f){};

    for (int k0 = 0; k0 < K; k0 += 32) {
        // A fragments: lane holds row m, halfs K = kbA..kbA+15 (contiguous)
        v16h a[2];
        {
            v8h lo0 = *(const v8h*)(arow0 + k0 + kbA);
            v8h hi0 = *(const v8h*)(arow0 + k0 + kbA + 8);
            v8h lo1 = *(const v8h*)(arow1 + k0 + kbA);
            v8h hi1 = *(const v8h*)(arow1 + k0 + kbA + 8);
            #pragma unroll
            for (int i = 0; i < 8; ++i) {
                a[0][i] = lo0[i]; a[0][i + 8] = hi0[i];
                a[1][i] = lo1[i]; a[1][i + 8] = hi1[i];
            }
        }
        // B fragments: lane holds column n, halfs K = kbB..kbB+15 (contiguous in Bt)
        v16h bf[4];
        #pragma unroll
        for (int j = 0; j < 4; ++j)
            bf[j] = *(const v16h*)(brow[j] + k0 + kbB);

        #pragma unroll
        for (int i = 0; i < 2; ++i)
            #pragma unroll
            for (int j = 0; j < 4; ++j)
                acc[i][j] = __builtin_amdgcn_wmma_f32_16x16x32_f16(
                                false, a[i], false, bf[j], (short)0, acc[i][j],
                                false, false);
    }

    // D layout: VGPR v, lane L -> m = v + 8*(L>>4), n = L&15
    const bool has_res = (residual != nullptr);
    #pragma unroll
    for (int i = 0; i < 2; ++i) {
        const int mbase = (mtile0 + i) * 16 + 8 * (lane >> 4);
        #pragma unroll
        for (int j = 0; j < 4; ++j) {
            const int ncol = (ntile0 + j) * 16 + (lane & 15);
            #pragma unroll
            for (int v = 0; v < 8; ++v) {
                size_t idx = (size_t)(mbase + v) * N + ncol;
                float val = acc[i][j][v];
                if (has_res) val += residual[idx];
                C[idx] = val;
            }
        }
    }
}

// ---------------------------------------------------------------------------
// 4) Causal depthwise conv (k=4) + bias + SiLU.  x_main = xz[:, 0:D_INNER].
// ---------------------------------------------------------------------------
__global__ __launch_bounds__(256)
void ssm_conv_silu_kernel(const float* __restrict__ xz,
                          const float* __restrict__ cw,
                          const float* __restrict__ cb,
                          float* __restrict__ xc)
{
    int idx = blockIdx.x * 256 + threadIdx.x;           // over ROWS*D_INNER
    if (idx >= ROWS * D_INNER) return;
    const int d = idx % D_INNER;
    const int t = (idx / D_INNER) % SEQ;
    const int b = idx / (D_INNER * SEQ);

    float acc = cb[d];
    #pragma unroll
    for (int j = 0; j < D_CONV; ++j) {
        int tt = t - (D_CONV - 1) + j;
        if (tt >= 0)
            acc += cw[d * D_CONV + j] * xz[((size_t)(b * SEQ + tt)) * (2 * D_INNER) + d];
    }
    xc[idx] = acc / (1.f + __expf(-acc));               // SiLU
}

// ---------------------------------------------------------------------------
// 5) x_proj: proj[row, 0:129] = xc[row, :] @ W[1536,129]   (plain f32)
// ---------------------------------------------------------------------------
__global__ __launch_bounds__(256)
void ssm_xproj_kernel(const float* __restrict__ xc,
                      const float* __restrict__ W,
                      float* __restrict__ proj)
{
    const int row = blockIdx.x;
    const int tid = threadIdx.x;
    __shared__ float sx[D_INNER];
    for (int k = tid; k < D_INNER; k += 256) sx[k] = xc[(size_t)row * D_INNER + k];
    __syncthreads();
    for (int j = tid; j < NPROJ; j += 256) {
        float s = 0.f;
        for (int k = 0; k < D_INNER; ++k) s += sx[k] * W[(size_t)k * NPROJ + j];
        proj[(size_t)row * NPROJ + j] = s;
    }
}

// ---------------------------------------------------------------------------
// 6) Selective scan: one thread per (b,d) channel, 64-state in VGPRs,
//    per-step B/C vectors broadcast via LDS.  Writes gated y as f16 (for GEMM3).
// ---------------------------------------------------------------------------
__global__ __launch_bounds__(128)
void ssm_scan_kernel(const float* __restrict__ xc,
                     const float* __restrict__ proj,
                     const float* __restrict__ xz,     // for z gate
                     const float* __restrict__ A_log,
                     const float* __restrict__ Dvec,
                     const float* __restrict__ dt_w,
                     const float* __restrict__ dt_b,
                     _Float16* __restrict__ y16)
{
    const int b   = blockIdx.y;
    const int d   = blockIdx.x * 128 + threadIdx.x;
    const int tid = threadIdx.x;

    __shared__ float sA[D_STATE], sB[D_STATE], sC[D_STATE];
    if (tid < D_STATE) sA[tid] = -__expf(A_log[tid]);

    float h[D_STATE];
    #pragma unroll
    for (int s = 0; s < D_STATE; ++s) h[s] = 0.f;

    const float dtw = dt_w[d];
    const float dtb = dt_b[d];
    const float Dd  = Dvec[d];

    for (int t = 0; t < SEQ; ++t) {
        const size_t row = (size_t)(b * SEQ + t);
        __syncthreads();                                // protect sB/sC reuse
        if (tid < D_STATE)            sB[tid]       = proj[row * NPROJ + tid];
        else if (tid < 2 * D_STATE)   sC[tid - 64]  = proj[row * NPROJ + 64 + (tid - 64)];
        __syncthreads();

        const float xv  = xc[row * D_INNER + d];
        const float arg = proj[row * NPROJ + (NPROJ - 1)] * dtw + dtb;
        const float dt  = (arg > 20.f) ? arg : log1pf(__expf(arg)); // softplus
        const float xdt = dt * xv;

        float y = 0.f;
        #pragma unroll
        for (int s = 0; s < D_STATE; ++s) {
            const float dA = __expf(sA[s] * dt);
            h[s] = dA * h[s] + xdt * sB[s];
            y += h[s] * sC[s];
        }
        y += Dd * xv;
        const float zv = xz[row * (2 * D_INNER) + D_INNER + d];
        y *= zv / (1.f + __expf(-zv));                  // * silu(z)
        y16[row * D_INNER + d] = (_Float16)y;
    }
}

// ---------------------------------------------------------------------------
// Launch
// ---------------------------------------------------------------------------
static inline size_t align256(size_t x) { return (x + 255) & ~(size_t)255; }

extern "C" void kernel_launch(void* const* d_in, const int* in_sizes, int n_in,
                              void* d_out, int out_size, void* d_ws, size_t ws_size,
                              hipStream_t stream)
{
    const float* x        = (const float*)d_in[0];   // (2,1024,768)
    const float* ln_w     = (const float*)d_in[1];
    const float* ln_b     = (const float*)d_in[2];
    const float* in_w     = (const float*)d_in[3];   // (768, 3072)
    const float* conv_w   = (const float*)d_in[4];   // (1536,1,4)
    const float* conv_b   = (const float*)d_in[5];
    const float* xproj_w  = (const float*)d_in[6];   // (1536,129)
    const float* A_log    = (const float*)d_in[7];   // (64)
    const float* Dvec     = (const float*)d_in[8];   // (1536)
    const float* dt_w     = (const float*)d_in[9];   // (1,1536)
    const float* dt_b     = (const float*)d_in[10];  // (1536)
    const float* out_w    = (const float*)d_in[11];  // (1536,768)
    float*       out      = (float*)d_out;           // (2,1024,768)

    // workspace carve-out (all 256B aligned)
    char* ws = (char*)d_ws;
    size_t off = 0;
    _Float16* h16  = (_Float16*)(ws + off); off = align256(off + (size_t)ROWS   * D_MODEL * 2);
    _Float16* w1t  = (_Float16*)(ws + off); off = align256(off + (size_t)(2*D_INNER) * D_MODEL * 2);
    float*    xz   = (float*)   (ws + off); off = align256(off + (size_t)ROWS   * 2 * D_INNER * 4);
    float*    xc   = (float*)   (ws + off); off = align256(off + (size_t)ROWS   * D_INNER * 4);
    float*    proj = (float*)   (ws + off); off = align256(off + (size_t)ROWS   * NPROJ   * 4);
    _Float16* y16  = (_Float16*)(ws + off); off = align256(off + (size_t)ROWS   * D_INNER * 2);
    _Float16* w3t  = (_Float16*)(ws + off); off = align256(off + (size_t)D_MODEL * D_INNER * 2);
    (void)ws_size; (void)in_sizes; (void)n_in; (void)out_size;

    // 1) LayerNorm -> f16
    ssm_ln_kernel<<<ROWS, 256, 0, stream>>>(x, ln_w, ln_b, h16);

    // 2) Weight transposes -> f16 N-major
    {
        int n1 = D_MODEL * 2 * D_INNER;
        ssm_wt_kernel<<<(n1 + 255) / 256, 256, 0, stream>>>(in_w,  w1t, D_MODEL, 2 * D_INNER);
        int n3 = D_INNER * D_MODEL;
        ssm_wt_kernel<<<(n3 + 255) / 256, 256, 0, stream>>>(out_w, w3t, D_INNER, D_MODEL);
    }

    // 3) GEMM1: xz = h @ in_proj_w     (2048x768 @ 768x3072), 32x64 per wave
    {
        dim3 grid(ROWS / 32, (2 * D_INNER) / 256), block(32, 4);
        ssm_wmma_gemm_kernel<<<grid, block, 0, stream>>>(h16, w1t, xz, nullptr,
                                                         ROWS, 2 * D_INNER, D_MODEL);
    }

    // 4) depthwise conv + SiLU
    ssm_conv_silu_kernel<<<(ROWS * D_INNER + 255) / 256, 256, 0, stream>>>(xz, conv_w, conv_b, xc);

    // 5) x_proj (B, C, dt_raw)
    ssm_xproj_kernel<<<ROWS, 256, 0, stream>>>(xc, xproj_w, proj);

    // 6) selective scan + gating -> y16
    {
        dim3 grid(D_INNER / 128, BATCH), block(128, 1);
        ssm_scan_kernel<<<grid, block, 0, stream>>>(xc, proj, xz, A_log, Dvec, dt_w, dt_b, y16);
    }

    // 7) GEMM3: out = y @ out_proj_w + residual   (2048x1536 @ 1536x768)
    {
        dim3 grid(ROWS / 32, D_MODEL / 256), block(32, 4);
        ssm_wmma_gemm_kernel<<<grid, block, 0, stream>>>(y16, w3t, out, x,
                                                         ROWS, D_MODEL, D_INNER);
    }
}
